// AS_Mamba_Block_36412732735926
// MI455X (gfx1250) — compile-verified, matrix operand order checked by reading, not run on hardware
//
#include <hip/hip_runtime.h>

// ---------------------------------------------------------------------------
// AS_Mamba_Block for MI455X (gfx1250, wave32, WMMA).
// - 1728 independent queries chunked 4x432 (28080 tokens/chunk; the whole
//   per-chunk working set ~117MB stays L2-resident on the 192MB L2).
// - All GEMMs via v_wmma_f32_16x16x32_bf16 with pre-transposed bf16 weights
//   (B^T [Npad x K], Npad % 64 == 0) so both fragments are contiguous 16B
//   vector loads and each wave computes a 16x64 slab with no divergence.
// - Selective scan stays in f32 VALU (transcendental-bound, not matrix work).
// ---------------------------------------------------------------------------

typedef __attribute__((ext_vector_type(16))) __bf16 v16bf;
typedef __attribute__((ext_vector_type(8)))  float  v8f;

#define HQ    36
#define WQ    48
#define HWQ   1728
#define SEQL  65
#define NCHUNK 432
#define NCHUNKS 4
#define TCHUNK (NCHUNK * SEQL)   // 28080 tokens, multiple of 16

__device__ __forceinline__ float bf2f(unsigned short u) {
    union { unsigned int i; float f; } x; x.i = ((unsigned int)u) << 16; return x.f;
}
__device__ __forceinline__ unsigned short f2bf(float f) {
    union { float f; unsigned int i; } x; x.f = f;
    unsigned int i = x.i;
    unsigned int r = i + 0x7FFFu + ((i >> 16) & 1u);   // round-to-nearest-even
    return (unsigned short)(r >> 16);
}

union Frag16 { uint4 u[2]; unsigned short s[16]; v16bf v; };

// ---------------------------------------------------------------------------
// Weight prep: W[K x N] f32 -> BT[Npad x K] bf16, rows n >= N zero-filled.
// BT-indexed so padding is uniform; one-shot, tiny.
// ---------------------------------------------------------------------------
__global__ void wt_kernel(const float* __restrict__ W, unsigned short* __restrict__ BT,
                          int K, int N, int Npad)
{
    int idx = blockIdx.x * 256 + threadIdx.x;
    if (idx >= Npad * K) return;
    int n = idx / K, k = idx % K;
    BT[idx] = (n < N) ? f2bf(W[(long long)k * N + n]) : (unsigned short)0;
}

// ---------------------------------------------------------------------------
// bf16 WMMA GEMM: Out[M x Npad] = A[M x K](bf16) * B, B given as
// BT[Npad x K] bf16. One wave computes a 16x64 output slab (four 16x16
// tiles sharing one A fragment). K stepped by 32.
// Requires M % 16 == 0, Npad % 64 == 0 -> no divergent control flow at all.
// MODE 0: store bf16   MODE 1: store f32   MODE 2: f32 accumulate (residual)
//
// 16-bit A 16x32 lane layout: lanes 0-15 row=r hold K {0..7,16..23},
// lanes 16-31 row=r hold K {8..15,24..31}; B mirrored (col across lanes),
// which with the BT layout is the same contiguous 16-element pattern.
// C/D: VGPR v -> row v (lanes 0-15) / v+8 (lanes 16-31), col = lane%16.
// ---------------------------------------------------------------------------
template <int MODE>
__global__ void __launch_bounds__(256)
wmma_gemm(const unsigned short* __restrict__ A, int lda,
          const unsigned short* __restrict__ BT,
          void* __restrict__ Out, int ldo,
          int K, int quadsN, int totalQuads)
{
    int wave = (blockIdx.x * 256 + threadIdx.x) >> 5;
    if (wave >= totalQuads) return;
    int lane = threadIdx.x & 31;
    int r    = lane & 15;
    int half = lane >> 4;
    int tm  = wave / quadsN;
    int tn  = (wave % quadsN) * 4;          // first of 4 N-tiles

    const unsigned short* arow = A  + (long long)(tm * 16 + r) * lda + (half ? 8 : 0);
    const unsigned short* bcol = BT + (long long)(tn * 16 + r) * K  + (half ? 8 : 0);
    const long long bstep = (long long)16 * K;   // next N-tile's column group

    v8f acc[4] = {{}, {}, {}, {}};
    for (int kb = 0; kb < K; kb += 32) {
        Frag16 a;
        a.u[0] = *(const uint4*)(arow + kb);
        a.u[1] = *(const uint4*)(arow + kb + 16);
#pragma unroll
        for (int u = 0; u < 4; ++u) {
            Frag16 b;
            b.u[0] = *(const uint4*)(bcol + u * bstep + kb);
            b.u[1] = *(const uint4*)(bcol + u * bstep + kb + 16);
            acc[u] = __builtin_amdgcn_wmma_f32_16x16x32_bf16(
                         false, a.v, false, b.v, (short)0, acc[u], false, false);
        }
    }

    long long obase = (long long)(tm * 16 + (half ? 8 : 0)) * ldo + r;
#pragma unroll
    for (int u = 0; u < 4; ++u) {
        long long ob = obase + (tn + u) * 16;
#pragma unroll
        for (int v = 0; v < 8; ++v) {
            long long oidx = ob + (long long)v * ldo;
            if (MODE == 0)      ((unsigned short*)Out)[oidx] = f2bf(acc[u][v]);
            else if (MODE == 1) ((float*)Out)[oidx] = acc[u][v];
            else                ((float*)Out)[oidx] += acc[u][v];
        }
    }
}

// ---------------------------------------------------------------------------
// Bilinear sampling + sequence assembly for one chunk.
// One block (256 thr) per query. Token 0 = query features, tokens 1..64 =
// bilinear samples of the target maps on the span-scaled 8x8 grid.
// ---------------------------------------------------------------------------
__global__ void __launch_bounds__(256)
sample_kernel(const float* __restrict__ fmq, const float* __restrict__ fgq,
              const float* __restrict__ fmt, const float* __restrict__ fgt,
              const float* __restrict__ flow, const float* __restrict__ sx,
              const float* __restrict__ sy,
              float* __restrict__ sm, float* __restrict__ sg, int q0)
{
    __shared__ int   so[64][4];
    __shared__ float sw[64][4];
    int ql = blockIdx.x;
    int n  = q0 + ql;
    int iy = n / WQ, ix = n % WQ;
    int tid = threadIdx.x;

    if (tid < 64) {
        int si = tid >> 3, sj = tid & 7;
        float lin_i = -1.f + 2.f * si / 7.f;       // base grid x component
        float lin_j = -1.f + 2.f * sj / 7.f;       // base grid y component
        float cx = -1.f + 2.f * ix / (WQ - 1.f);
        float cy = -1.f + 2.f * iy / (HQ - 1.f);
        float ctrx = cx + 2.f * flow[n * 2 + 0] / (WQ - 1.f);
        float ctry = cy + 2.f * flow[n * 2 + 1] / (HQ - 1.f);
        float scx = 2.f * sx[n] / (WQ - 1.f);
        float scy = 2.f * sy[n] / (HQ - 1.f);
        float gx = lin_i * scx + ctrx;
        float gy = lin_j * scy + ctry;
        float gxp = (gx + 1.f) * 0.5f * (WQ - 1.f);
        float gyp = (gy + 1.f) * 0.5f * (HQ - 1.f);
        float x0f = floorf(gxp), y0f = floorf(gyp);
        float wx1 = gxp - x0f, wx0 = 1.f - wx1;
        float wy1 = gyp - y0f, wy0 = 1.f - wy1;
        int x0 = (int)x0f, y0 = (int)y0f;
        int x1 = x0 + 1,   y1 = y0 + 1;
        float v00 = (x0 >= 0 && x0 <= WQ - 1 && y0 >= 0 && y0 <= HQ - 1) ? 1.f : 0.f;
        float v10 = (x1 >= 0 && x1 <= WQ - 1 && y0 >= 0 && y0 <= HQ - 1) ? 1.f : 0.f;
        float v01 = (x0 >= 0 && x0 <= WQ - 1 && y1 >= 0 && y1 <= HQ - 1) ? 1.f : 0.f;
        float v11 = (x1 >= 0 && x1 <= WQ - 1 && y1 >= 0 && y1 <= HQ - 1) ? 1.f : 0.f;
        int cx0 = min(max(x0, 0), WQ - 1), cx1 = min(max(x1, 0), WQ - 1);
        int cy0 = min(max(y0, 0), HQ - 1), cy1 = min(max(y1, 0), HQ - 1);
        so[tid][0] = cy0 * WQ + cx0;  sw[tid][0] = wx0 * wy0 * v00;
        so[tid][1] = cy0 * WQ + cx1;  sw[tid][1] = wx1 * wy0 * v10;
        so[tid][2] = cy1 * WQ + cx0;  sw[tid][2] = wx0 * wy1 * v01;
        so[tid][3] = cy1 * WQ + cx1;  sw[tid][3] = wx1 * wy1 * v11;
    }
    __syncthreads();

    // token 0: query features
    for (int c = tid; c < 320; c += 256) {
        if (c < 256) sm[(long long)ql * SEQL * 256 + c] = fmq[(long long)c * HWQ + n];
        else         sg[(long long)ql * SEQL * 64 + (c - 256)] = fgq[(long long)(c - 256) * HWQ + n];
    }
    // tokens 1..64: bilinear samples
    for (int idx = tid; idx < 64 * 320; idx += 256) {
        int s = idx / 320, c = idx % 320;
        const int*   o = so[s];
        const float* w = sw[s];
        if (c < 256) {
            const float* f = fmt + (long long)c * HWQ;
            float v = f[o[0]] * w[0] + f[o[1]] * w[1] + f[o[2]] * w[2] + f[o[3]] * w[3];
            sm[((long long)ql * SEQL + 1 + s) * 256 + c] = v;
        } else {
            int cg = c - 256;
            const float* f = fgt + (long long)cg * HWQ;
            float v = f[o[0]] * w[0] + f[o[1]] * w[1] + f[o[2]] * w[2] + f[o[3]] * w[3];
            sg[((long long)ql * SEQL + 1 + s) * 64 + cg] = v;
        }
    }
}

// ---------------------------------------------------------------------------
// RMSNorm: X_bf16[t][c] = S[t][c] * rsqrt(mean(S^2)+1e-5) * w[c]. block = C.
// ---------------------------------------------------------------------------
__global__ void rms_kernel(const float* __restrict__ S, const float* __restrict__ w,
                           unsigned short* __restrict__ X, int C)
{
    __shared__ float red[256];
    long long t = blockIdx.x;
    int c = threadIdx.x;
    float v = S[t * C + c];
    red[c] = v * v;
    __syncthreads();
    for (int off = C >> 1; off > 0; off >>= 1) {
        if (c < off) red[c] += red[c + off];
        __syncthreads();
    }
    float scale = rsqrtf(red[0] / C + 1e-5f);
    X[t * C + c] = f2bf(v * scale * w[c]);
}

// ---------------------------------------------------------------------------
// Fused dt-projection + softplus + selective scan + SiLU gating.
// One lane owns one d_in channel j of one query, keeps h[16] in VGPRs and
// walks the 65-token sequence. Writes gated y in bf16 for the out-GEMM.
// dbc rows are 64 floats (48 used + 16 pad) = 256B, nicely aligned.
// ---------------------------------------------------------------------------
__global__ void __launch_bounds__(256)
scan_kernel(const float* __restrict__ dbc,
            const unsigned short* __restrict__ xz, int ldxz,
            const unsigned short* __restrict__ cond,
            const float* __restrict__ dtw,   // [16][DIN]
            const float* __restrict__ dtb,   // [DIN]
            const float* __restrict__ Alog,  // [DIN][16]
            const float* __restrict__ Dp,    // [DIN]
            unsigned short* __restrict__ ybf,
            int DIN, int nquery)
{
    int gw   = (blockIdx.x * 256 + threadIdx.x) >> 5;
    int lane = threadIdx.x & 31;
    int gpq  = DIN >> 5;
    int q = gw / gpq;
    if (q >= nquery) return;
    int j = (gw % gpq) * 32 + lane;

    float wdt[16], Ae[16], h[16];
#pragma unroll
    for (int s = 0; s < 16; ++s) {
        wdt[s] = dtw[s * DIN + j];
        Ae[s]  = -__expf(Alog[j * 16 + s]);
        h[s]   = 0.f;
    }
    float dtbj = dtb[j];
    float Dj   = Dp[j];

    for (int tt = 0; tt < SEQL; ++tt) {
        long long t = (long long)q * SEQL + tt;
        const float* dr = dbc + t * 64;
        float dot = dtbj + bf2f(cond[t * DIN + j]);
#pragma unroll
        for (int k = 0; k < 16; ++k) dot += dr[k] * wdt[k];
        float dts = (dot > 20.f) ? dot : log1pf(__expf(dot));   // softplus
        float xt  = bf2f(xz[t * ldxz + j]);
        float dx  = dts * xt;
        float acc = 0.f;
#pragma unroll
        for (int s = 0; s < 16; ++s) {
            float hs = h[s] * __expf(dts * Ae[s]) + dx * dr[16 + s];
            h[s] = hs;
            acc += hs * dr[32 + s];
        }
        float zz = bf2f(xz[t * ldxz + DIN + j]);
        float y  = (acc + Dj * xt) * (zz / (1.f + __expf(-zz)));  // * silu(z)
        ybf[t * DIN + j] = f2bf(y);
    }
}

// ---------------------------------------------------------------------------
// out[c][n] = q[c][n] + seq[query][token0][c]   (NCHW output layout)
// ---------------------------------------------------------------------------
__global__ void final_kernel(const float* __restrict__ q, const float* __restrict__ seq,
                             float* __restrict__ out, int q0, int C)
{
    int idx = blockIdx.x * 256 + threadIdx.x;
    if (idx >= NCHUNK * C) return;
    int ql = idx / C, c = idx % C;
    int n = q0 + ql;
    out[(long long)c * HWQ + n] = q[(long long)c * HWQ + n]
                                + seq[((long long)ql * SEQL) * C + c];
}

// ---------------------------------------------------------------------------
static void launch_gemm(int mode, const unsigned short* A, int lda,
                        const unsigned short* BT, void* Out, int ldo,
                        int M, int K, int Npad, hipStream_t s)
{
    int quadsN = Npad / 64;                 // Npad % 64 == 0 by construction
    int tilesM = M / 16;                    // M % 16 == 0 by construction
    int total  = tilesM * quadsN;
    int blocks = (total + 7) / 8;
    if (mode == 0)      wmma_gemm<0><<<blocks, 256, 0, s>>>(A, lda, BT, Out, ldo, K, quadsN, total);
    else if (mode == 1) wmma_gemm<1><<<blocks, 256, 0, s>>>(A, lda, BT, Out, ldo, K, quadsN, total);
    else                wmma_gemm<2><<<blocks, 256, 0, s>>>(A, lda, BT, Out, ldo, K, quadsN, total);
}

extern "C" void kernel_launch(void* const* d_in, const int* in_sizes, int n_in,
                              void* d_out, int out_size, void* d_ws, size_t ws_size,
                              hipStream_t stream)
{
    (void)in_sizes; (void)n_in; (void)out_size; (void)ws_size;

    const float* fm[2]   = {(const float*)d_in[0], (const float*)d_in[1]};
    const float* fg[2]   = {(const float*)d_in[2], (const float*)d_in[3]};
    const float* flow[2] = {(const float*)d_in[4], (const float*)d_in[5]};
    const float* sx[2]   = {(const float*)d_in[6], (const float*)d_in[8]};
    const float* sy[2]   = {(const float*)d_in[7], (const float*)d_in[9]};
    const float* m_in    = (const float*)d_in[10];
    const float* m_xproj = (const float*)d_in[11];
    const float* m_dt    = (const float*)d_in[12];
    const float* m_dtb   = (const float*)d_in[13];
    const float* m_Alog  = (const float*)d_in[14];
    const float* m_D     = (const float*)d_in[15];
    const float* m_outw  = (const float*)d_in[16];
    const float* m_gcond = (const float*)d_in[17];
    const float* m_norm  = (const float*)d_in[18];
    const float* g_in    = (const float*)d_in[19];
    const float* g_xproj = (const float*)d_in[20];
    const float* g_dt    = (const float*)d_in[21];
    const float* g_dtb   = (const float*)d_in[22];
    const float* g_Alog  = (const float*)d_in[23];
    const float* g_D     = (const float*)d_in[24];
    const float* g_outw  = (const float*)d_in[25];
    const float* g_mcond = (const float*)d_in[26];
    const float* g_norm  = (const float*)d_in[27];

    // workspace carve-up (per-chunk activation buffers + bf16 weight cache)
    char* ws = (char*)d_ws;
    size_t off = 0;
    auto carve = [&](size_t bytes) {
        char* p = ws + off;
        off += (bytes + 255) & ~(size_t)255;
        return p;
    };
    float*          sm      = (float*)carve((size_t)TCHUNK * 256 * 4);
    float*          sg      = (float*)carve((size_t)TCHUNK * 64 * 4);
    unsigned short* xm_bf   = (unsigned short*)carve((size_t)TCHUNK * 256 * 2);
    unsigned short* xg_bf   = (unsigned short*)carve((size_t)TCHUNK * 64 * 2);
    unsigned short* cond_bf = (unsigned short*)carve((size_t)TCHUNK * 256 * 2);
    unsigned short* xz_bf   = (unsigned short*)carve((size_t)TCHUNK * 512 * 2);
    float*          dbc     = (float*)carve((size_t)TCHUNK * 64 * 4);   // 48 used + 16 pad
    unsigned short* y_bf    = (unsigned short*)carve((size_t)TCHUNK * 256 * 2);

    // bf16 transposed weight cache: BT[Npad x K] per (matrix, depth)
    unsigned short* bt_m_in    = (unsigned short*)carve((size_t)2 * 512 * 256 * 2);
    unsigned short* bt_m_gcond = (unsigned short*)carve((size_t)2 * 256 * 64 * 2);
    unsigned short* bt_m_xproj = (unsigned short*)carve((size_t)2 * 64 * 256 * 2);  // padded 48->64
    unsigned short* bt_m_outw  = (unsigned short*)carve((size_t)2 * 256 * 256 * 2);
    unsigned short* bt_g_mcond = (unsigned short*)carve((size_t)2 * 64 * 256 * 2);
    unsigned short* bt_g_in    = (unsigned short*)carve((size_t)2 * 128 * 64 * 2);
    unsigned short* bt_g_xproj = (unsigned short*)carve((size_t)2 * 64 * 64 * 2);   // padded 48->64
    unsigned short* bt_g_outw  = (unsigned short*)carve((size_t)2 * 64 * 64 * 2);

    // one-shot weight convert+transpose+pad (runs every launch; deterministic)
    struct { const float* W; unsigned short* BT; int K, N, Npad; } wt[8] = {
        { m_in,    bt_m_in,    256, 512, 512 },
        { m_gcond, bt_m_gcond, 64,  256, 256 },
        { m_xproj, bt_m_xproj, 256, 48,  64  },
        { m_outw,  bt_m_outw,  256, 256, 256 },
        { g_mcond, bt_g_mcond, 256, 64,  64  },
        { g_in,    bt_g_in,    64,  128, 128 },
        { g_xproj, bt_g_xproj, 64,  48,  64  },
        { g_outw,  bt_g_outw,  64,  64,  64  },
    };
    for (int i = 0; i < 8; ++i) {
        int elems = wt[i].Npad * wt[i].K;
        for (int d = 0; d < 2; ++d) {
            wt_kernel<<<(elems + 255) / 256, 256, 0, stream>>>(
                wt[i].W + (size_t)d * wt[i].K * wt[i].N,
                wt[i].BT + (size_t)d * elems,
                wt[i].K, wt[i].N, wt[i].Npad);
        }
    }

    float* outp = (float*)d_out;
    const size_t OFF_M[2] = {0, (size_t)256 * HWQ};
    const size_t OFF_G[2] = {(size_t)2 * 256 * HWQ, (size_t)2 * 256 * HWQ + (size_t)64 * HWQ};

    for (int dir = 0; dir < 2; ++dir) {
        const float* fmq = fm[dir];
        const float* fgq = fg[dir];
        const float* fmt = fm[1 - dir];
        const float* fgt = fg[1 - dir];
        float* out_m = outp + OFF_M[dir];
        float* out_g = outp + OFF_G[dir];

        for (int ch = 0; ch < NCHUNKS; ++ch) {
            int q0 = ch * NCHUNK;
            sample_kernel<<<NCHUNK, 256, 0, stream>>>(fmq, fgq, fmt, fgt,
                                                      flow[dir], sx[dir], sy[dir],
                                                      sm, sg, q0);
            for (int d = 0; d < 2; ++d) {
                rms_kernel<<<TCHUNK, 256, 0, stream>>>(sm, m_norm + d * 256, xm_bf, 256);
                rms_kernel<<<TCHUNK, 64, 0, stream>>>(sg, g_norm + d * 64, xg_bf, 64);

                // ---- m path ----
                launch_gemm(0, xg_bf, 64,  bt_m_gcond + (size_t)d * 256 * 64,  cond_bf, 256, TCHUNK, 64,  256, stream);
                launch_gemm(0, xm_bf, 256, bt_m_in    + (size_t)d * 512 * 256, xz_bf,   512, TCHUNK, 256, 512, stream);
                launch_gemm(1, xz_bf, 512, bt_m_xproj + (size_t)d * 64 * 256,  dbc,     64,  TCHUNK, 256, 64,  stream);
                scan_kernel<<<(NCHUNK * 8 * 32) / 256, 256, 0, stream>>>(
                    dbc, xz_bf, 512, cond_bf,
                    m_dt + d * 16 * 256, m_dtb + d * 256,
                    m_Alog + d * 256 * 16, m_D + d * 256,
                    y_bf, 256, NCHUNK);
                launch_gemm(2, y_bf, 256, bt_m_outw + (size_t)d * 256 * 256, sm, 256, TCHUNK, 256, 256, stream);

                // ---- g path (uses xm_bf/xg_bf captured before sm update) ----
                launch_gemm(0, xm_bf, 256, bt_g_mcond + (size_t)d * 64 * 256, cond_bf, 64,  TCHUNK, 256, 64,  stream);
                launch_gemm(0, xg_bf, 64,  bt_g_in    + (size_t)d * 128 * 64, xz_bf,   128, TCHUNK, 64,  128, stream);
                launch_gemm(1, xz_bf, 128, bt_g_xproj + (size_t)d * 64 * 64,  dbc,     64,  TCHUNK, 64,  64,  stream);
                scan_kernel<<<(NCHUNK * 2 * 32) / 256, 256, 0, stream>>>(
                    dbc, xz_bf, 128, cond_bf,
                    g_dt + d * 16 * 64, g_dtb + d * 64,
                    g_Alog + d * 64 * 16, g_D + d * 64,
                    y_bf, 64, NCHUNK);
                launch_gemm(2, y_bf, 64, bt_g_outw + (size_t)d * 64 * 64, sg, 64, TCHUNK, 64, 64, stream);
            }
            final_kernel<<<(NCHUNK * 256 + 255) / 256, 256, 0, stream>>>(fmq, sm, out_m, q0, 256);
            final_kernel<<<(NCHUNK * 64 + 255) / 256, 256, 0, stream>>>(fgq, sg, out_g, q0, 64);
        }
    }
}
